// NAFBlock_73761768341770
// MI455X (gfx1250) — compile-verified, matrix operand order checked by reading, not run on hardware
//
#include <hip/hip_runtime.h>
#include <hip/hip_bf16.h>

typedef __attribute__((ext_vector_type(16))) __bf16 v16bf;
typedef __attribute__((ext_vector_type(8)))  __bf16 v8bf;
typedef __attribute__((ext_vector_type(8)))  float  v8f;

typedef unsigned int tdm_g0_t __attribute__((ext_vector_type(4)));
typedef int          tdm_g1_t __attribute__((ext_vector_type(8)));
typedef int          tdm_g2_t __attribute__((ext_vector_type(4)));

namespace {
constexpr int  kB = 8, kH = 256, kW = 256, kC = 64, kHid = 128, kHid2 = 256;
constexpr float kEps = 1e-6f;
}

// Pack two floats as adjacent bf16 into a u32 (for 4-byte LDS stores).
__device__ __forceinline__ unsigned pack2(float a, float b) {
  unsigned short ha = __builtin_bit_cast(unsigned short, (__bf16)a);
  unsigned short hb = __builtin_bit_cast(unsigned short, (__bf16)b);
  return (unsigned)ha | ((unsigned)hb << 16);
}

// Fragment swizzle: channel k (0..31 within a 32-wide k-step) -> (lane-half, elem)
//   half = (k>>3)&1 ; e = ((k>>4)<<3)|(k&7)      (inverse of the ISA A/B layout)

// gfx1250 async global->LDS copy (ASYNCcnt path).  LDS address = addr[31:0]
// of the flat pointer (ISA 10.2: the LDS aperture keeps the byte offset in
// the low 32 bits of the flat address).
__device__ __forceinline__ void async_copy16(void* ldsDst, const void* gSrc) {
  asm volatile("global_load_async_to_lds_b128 %0, %1, off"
               :: "v"((unsigned)(unsigned long long)ldsDst),
                  "v"((unsigned long long)gSrc)
               : "memory");
}
__device__ __forceinline__ void wait_async() {
  asm volatile("s_wait_asynccnt 0x0" ::: "memory");
}

// ---------------------------------------------------------------------------
__global__ void k_zero(float* __restrict__ p, int n) {
  int i = blockIdx.x * blockDim.x + threadIdx.x;
  if (i < n) p[i] = 0.f;
}

// ---------------------------------------------------------------------------
// Kernel 1: LN1 + pw1 (64 -> 256).  One wave = 16 pixels; block = 8 waves.
// Output t1T layout: [tile][n(256)][m(16)] bf16 -> b128 epilogue stores.
// ---------------------------------------------------------------------------
__global__ __launch_bounds__(256) void k_ln1_pw1(
    const float* __restrict__ x, const float* __restrict__ g1,
    const float* __restrict__ b1n, const float* __restrict__ w1,
    const float* __restrict__ b1, __bf16* __restrict__ t1T) {
  __shared__ __bf16 wSw[2 * 16 * 32 * 16];  // 32 KB swizzled B fragments
  __shared__ float  bLds[kHid2];
  __shared__ __bf16 aSw[8 * 2 * 32 * 16];   // 16 KB per-wave A fragments

  const int tid = threadIdx.x, wave = tid >> 5, lane = tid & 31;

  for (int i = tid; i < kC * kHid2; i += 256) {  // swizzle pw1_w
    const int kk = i >> 8, n = i & 255;
    const int t = kk >> 5, k = kk & 31;
    const int hf = (k >> 3) & 1, e = ((k >> 4) << 3) | (k & 7);
    wSw[((t * 16 + (n >> 4)) * 32 + hf * 16 + (n & 15)) * 16 + e] = (__bf16)w1[i];
  }
  for (int i = tid; i < kHid2; i += 256) bLds[i] = b1[i];

  const long pix0 = ((long)blockIdx.x * 8 + wave) * 16;
  __builtin_prefetch(x + pix0 * kC, 0, 3);

  {  // LN over 64 ch (2 lanes/pixel), write A fragments swizzled
    const int p = lane >> 1, c0 = (lane & 1) * 32;
    const float* xp = x + (pix0 + p) * kC + c0;
    float v[32], s = 0.f, q = 0.f;
#pragma unroll
    for (int c = 0; c < 32; ++c) { float t = xp[c]; v[c] = t; s += t; q += t * t; }
    s += __shfl_xor(s, 1, 32);
    q += __shfl_xor(q, 1, 32);
    const float mean = s * (1.f / kC);
    const float rstd = rsqrtf(q * (1.f / kC) - mean * mean + kEps);
#pragma unroll
    for (int c = 0; c < 32; c += 2) {
      const int cg = c0 + c;
      const float f0 = (v[c] - mean) * rstd * g1[cg] + b1n[cg];
      const float f1 = (v[c + 1] - mean) * rstd * g1[cg + 1] + b1n[cg + 1];
      const int t = cg >> 5, k = cg & 31;
      const int hf = (k >> 3) & 1, e = ((k >> 4) << 3) | (k & 7);
      *(unsigned*)&aSw[((wave * 2 + t) * 32 + hf * 16 + p) * 16 + e] = pack2(f0, f1);
    }
  }
  __syncthreads();

  const v16bf a0 = *(const v16bf*)&aSw[((wave * 2 + 0) * 32 + lane) * 16];
  const v16bf a1 = *(const v16bf*)&aSw[((wave * 2 + 1) * 32 + lane) * 16];

  v8f acc[16];
#pragma unroll
  for (int nt = 0; nt < 16; ++nt)
#pragma unroll
    for (int j = 0; j < 8; ++j) acc[nt][j] = 0.f;

#pragma unroll
  for (int nt = 0; nt < 16; ++nt) {
    const v16bf b0 = *(const v16bf*)&wSw[((0 * 16 + nt) * 32 + lane) * 16];
    acc[nt] = __builtin_amdgcn_wmma_f32_16x16x32_bf16(
        false, a0, false, b0, (short)0, acc[nt], false, false);
    const v16bf b1f = *(const v16bf*)&wSw[((1 * 16 + nt) * 32 + lane) * 16];
    acc[nt] = __builtin_amdgcn_wmma_f32_16x16x32_bf16(
        false, a1, false, b1f, (short)0, acc[nt], false, false);
  }

  const int hf = lane >> 4, l15 = lane & 15;
  const long tile = (long)blockIdx.x * 8 + wave;
#pragma unroll
  for (int nt = 0; nt < 16; ++nt) {
    const int n = nt * 16 + l15;
    const float bias = bLds[n];
    v8bf pk;
#pragma unroll
    for (int j = 0; j < 8; ++j) pk[j] = (__bf16)(acc[nt][j] + bias);
    *(v8bf*)&t1T[(tile * kHid2 + n) * 16 + hf * 8] = pk;  // 8 contiguous m
  }
}

// ---------------------------------------------------------------------------
// Kernel 2: depthwise 3x3 + bias + SimpleGate -> y (bf16, pixel-major),
// GAP partials via wave-reduce + LDS/global atomics.
// Halo tile staged two ways (both CDNA5 DMA paths):
//   low gate-half  -> TDM tensor_load_to_lds (TENSORcnt), 3 descriptors/chunk
//   high gate-half -> global_load_async_to_lds_b128 (ASYNCcnt)
// LDS tile layout (both halves): [twn(3)][row(18)][ch(16)][m(16)] bf16.
// ---------------------------------------------------------------------------
__global__ __launch_bounds__(256) void k_dw_gate_gap(
    const __bf16* __restrict__ t1T, const float* __restrict__ dww,
    const float* __restrict__ dwb, __bf16* __restrict__ y,
    float* __restrict__ gap) {
  __shared__ __bf16 tLds[2 * 3 * 18 * 16 * 16];  // 54 KB (two halves)
  __shared__ float  dwC[9 * 32];                 // per-chunk dw weights
  __shared__ float  dbC[32];                     // per-chunk dw bias
  __shared__ float  gapLds[kHid];

  const int tid = threadIdx.x, wave = tid >> 5, lane = tid & 31;
  constexpr int kHiOff = 3 * 18 * 16 * 16;       // 13824 elements

  if (tid < kHid) gapLds[tid] = 0.f;

  const int tw = blockIdx.x & 15, th = (blockIdx.x >> 4) & 15, b = blockIdx.x >> 8;
  const int py = tid >> 4, px = tid & 15;
  const int hh = th * 16 + py, ww = tw * 16 + px;
  const long pix = ((long)b * kH + hh) * kW + ww;

  // Vertical halo clamping (uniform per block): LDS row r <=> hy = th*16-1+r.
  const int skipTop = (th == 0) ? 1 : 0;
  const int nRows   = 18 - skipTop - ((th == 15) ? 1 : 0);
  const int hy0     = th * 16 - 1 + skipTop;

  for (int cc = 0; cc < 8; ++cc) {
    __syncthreads();  // previous chunk fully consumed

    // per-chunk dw weights/bias: channels cc*16..+15 (low) and +128 (high)
    for (int i = tid; i < 9 * 32; i += 256) {
      const int kk = i >> 5, cl = i & 31;
      const int ch = (cl < 16) ? (cc * 16 + cl) : (128 + cc * 16 + (cl - 16));
      dwC[i] = dww[kk * kHid2 + ch];
    }
    if (tid < 32) {
      const int ch = (tid < 16) ? (cc * 16 + tid) : (128 + cc * 16 + (tid - 16));
      dbC[tid] = dwb[ch];
    }

    // ---- high half: async b128 copies (clamped; bad slots masked by taps)
    for (int i = tid; i < 16 * 18 * 3 * 2; i += 256) {
      const int cl = i / 108, rem = i - cl * 108;
      const int r = rem / 6, rem2 = rem - r * 6;
      const int twn_s = rem2 >> 1, g = rem2 & 1;
      const int ch = 128 + cc * 16 + cl;
      int hy = th * 16 - 1 + r;
      hy = hy < 0 ? 0 : (hy > kH - 1 ? kH - 1 : hy);
      int tg = tw - 1 + twn_s;
      tg = tg < 0 ? 0 : (tg > 15 ? 15 : tg);
      const int mb = g * 8;
      const long tileIdx = ((long)b * kH + hy) * 16 + tg;
      async_copy16(&tLds[kHiOff + ((twn_s * 18 + r) * 16 + cl) * 16 + mb],
                   &t1T[(tileIdx * kHid2 + ch) * 16 + mb]);
    }

    // ---- low half: Tensor Data Mover, one D# per w-tile slot
    if (wave == 0) {
      const int ch0 = cc * 16;
#pragma unroll
      for (int twn_s = 0; twn_s < 3; ++twn_s) {
        const int tg = tw - 1 + twn_s;
        if (tg < 0 || tg > 15) continue;  // edge slot: taps masked anyway
        const long tileIdx = ((long)b * kH + hy0) * 16 + tg;
        const unsigned long long ga =
            (unsigned long long)(const void*)(t1T + ((size_t)tileIdx * kHid2 + ch0) * 16);
        const unsigned ldsa = (unsigned)(unsigned long long)(const void*)
            &tLds[((twn_s * 18 + skipTop) * 16) * 16];
        // D# group0: count=1 | lds_addr | global_addr | type=2 (image)
        tdm_g0_t g0;
        g0[0] = 1u;
        g0[1] = ldsa;
        g0[2] = (unsigned)ga;
        g0[3] = (unsigned)((ga >> 32) & 0x1FFFFFFu) | 0x80000000u;
        // group1: data_size=2B; tile 16(m) x 16(ch) x nRows(hy);
        //         strides: ch=16 elems, hy=65536 elems; tensor dims huge.
        const unsigned TD = 0x40000000u;
        tdm_g1_t g1;
        g1[0] = (int)0x00010000u;
        g1[1] = (int)((TD & 0xFFFFu) << 16);
        g1[2] = (int)((TD >> 16) | ((TD & 0xFFFFu) << 16));
        g1[3] = (int)((TD >> 16) | (16u << 16));          // tile_dim0 = 16
        g1[4] = (int)(16u | ((unsigned)nRows << 16));     // tile_dim1, tile_dim2
        g1[5] = 16;                                        // dim0_stride lo
        g1[6] = 0;                                         // dim0_stride hi | dim1_stride lo
        g1[7] = 1;                                         // dim1_stride = 65536
        tdm_g2_t g2;
        g2[0] = (int)0x40000000u; g2[1] = 0; g2[2] = 0; g2[3] = 0;
        tdm_g2_t g3;
        g3[0] = 0; g3[1] = 0; g3[2] = 0; g3[3] = 0;
#if __clang_major__ >= 23
        tdm_g1_t zpad = (tdm_g1_t)0;
        __builtin_amdgcn_tensor_load_to_lds(g0, g1, g2, g3, zpad, 0);
#else
        __builtin_amdgcn_tensor_load_to_lds(g0, g1, g2, g3, 0);
#endif
      }
      __builtin_amdgcn_s_wait_tensorcnt((short)0);
    }
    wait_async();
    __syncthreads();

    v8bf y0, y1;
#pragma unroll
    for (int cl = 0; cl < 16; ++cl) {
      float v1 = dbC[cl], v2 = dbC[16 + cl];
#pragma unroll
      for (int dy = -1; dy <= 1; ++dy) {
        const int hy = hh + dy;
        if (hy < 0 || hy >= kH) continue;
        const int ro = py + 1 + dy;
#pragma unroll
        for (int dx = -1; dx <= 1; ++dx) {
          const int wx = ww + dx;
          if (wx < 0 || wx >= kW) continue;
          const int kk = (dy + 1) * 3 + (dx + 1);
          const int wrel = px + dx + 16;           // strip-relative w
          const int tl = wrel >> 4, m = wrel & 15;
          const int base = ((tl * 18 + ro) * 16 + cl) * 16 + m;
          v1 += dwC[kk * 32 + cl]      * (float)tLds[base];
          v2 += dwC[kk * 32 + 16 + cl] * (float)tLds[kHiOff + base];
        }
      }
      const float yv = v1 * v2;  // SimpleGate
      if (cl < 8) y0[cl] = (__bf16)yv; else y1[cl - 8] = (__bf16)yv;
      float s = yv;  // wave-reduce for GAP
      s += __shfl_xor(s, 1, 32);  s += __shfl_xor(s, 2, 32);
      s += __shfl_xor(s, 4, 32);  s += __shfl_xor(s, 8, 32);
      s += __shfl_xor(s, 16, 32);
      if (lane == 0) atomicAdd(&gapLds[cc * 16 + cl], s);
    }
    *(v8bf*)&y[pix * kHid + cc * 16] = y0;
    *(v8bf*)&y[pix * kHid + cc * 16 + 8] = y1;
  }
  __syncthreads();
  if (tid < kHid) atomicAdd(&gap[b * kHid + tid], gapLds[tid]);
}

// ---------------------------------------------------------------------------
// Kernel 3: SCA — w[b,o] = sca_b[o] + sum_c mean_y[b,c]*sca_w[c,o]
// ---------------------------------------------------------------------------
__global__ __launch_bounds__(128) void k_sca(
    const float* __restrict__ gap, const float* __restrict__ scaw,
    const float* __restrict__ scab, float* __restrict__ wbuf) {
  const int o = threadIdx.x;
  for (int b = 0; b < kB; ++b) {
    float acc = scab[o];
    for (int c = 0; c < kHid; ++c)
      acc += gap[b * kHid + c] * (1.f / (kH * kW)) * scaw[c * kHid + o];
    wbuf[b * kHid + o] = acc;
  }
}

// ---------------------------------------------------------------------------
// Kernel 4: channel attention + pw2 (128 -> 64) + residual -> x1 (f32)
// ---------------------------------------------------------------------------
__global__ __launch_bounds__(256) void k_attn_pw2_res(
    const __bf16* __restrict__ y, const float* __restrict__ wbuf,
    const float* __restrict__ w2, const float* __restrict__ b2,
    const float* __restrict__ x, const float* __restrict__ beta,
    float* __restrict__ x1) {
  __shared__ __bf16 wSw[4 * 4 * 32 * 16];   // 16 KB
  __shared__ __bf16 aSw[8 * 4 * 32 * 16];   // 32 KB
  __shared__ float  bLds[kC], betaLds[kC];

  const int tid = threadIdx.x, wave = tid >> 5, lane = tid & 31;

  for (int i = tid; i < kHid * kC; i += 256) {  // swizzle pw2_w
    const int kk = i >> 6, n = i & 63;
    const int t = kk >> 5, k = kk & 31;
    const int hf = (k >> 3) & 1, e = ((k >> 4) << 3) | (k & 7);
    wSw[((t * 4 + (n >> 4)) * 32 + hf * 16 + (n & 15)) * 16 + e] = (__bf16)w2[i];
  }
  if (tid < kC) { bLds[tid] = b2[tid]; betaLds[tid] = beta[tid]; }

  const long pix0 = ((long)blockIdx.x * 8 + wave) * 16;
  const int  b    = (int)(pix0 >> 16);

  {  // y tile (b128 loads), channel attention, A-fragment swizzle
    const int p = lane >> 1, c0 = (lane & 1) * 64;
    const __bf16* yp = y + (pix0 + p) * kHid + c0;
    const float*  wp = wbuf + b * kHid + c0;
    v8bf yraw[8];
#pragma unroll
    for (int q = 0; q < 8; ++q) yraw[q] = *(const v8bf*)(yp + q * 8);
#pragma unroll
    for (int c = 0; c < 64; c += 2) {
      const int cg = c0 + c;
      const float f0 = (float)yraw[c >> 3][c & 7] * wp[c];
      const float f1 = (float)yraw[(c + 1) >> 3][(c + 1) & 7] * wp[c + 1];
      const int t = cg >> 5, k = cg & 31;
      const int hf = (k >> 3) & 1, e = ((k >> 4) << 3) | (k & 7);
      *(unsigned*)&aSw[((wave * 4 + t) * 32 + hf * 16 + p) * 16 + e] = pack2(f0, f1);
    }
  }
  __syncthreads();

  v8f acc[4];
#pragma unroll
  for (int nt = 0; nt < 4; ++nt)
#pragma unroll
    for (int j = 0; j < 8; ++j) acc[nt][j] = 0.f;

#pragma unroll
  for (int t = 0; t < 4; ++t) {
    const v16bf a = *(const v16bf*)&aSw[((wave * 4 + t) * 32 + lane) * 16];
#pragma unroll
    for (int nt = 0; nt < 4; ++nt) {
      const v16bf bf = *(const v16bf*)&wSw[((t * 4 + nt) * 32 + lane) * 16];
      acc[nt] = __builtin_amdgcn_wmma_f32_16x16x32_bf16(
          false, a, false, bf, (short)0, acc[nt], false, false);
    }
  }

  const int hf = lane >> 4, l15 = lane & 15;
#pragma unroll
  for (int nt = 0; nt < 4; ++nt) {
    const int n = nt * 16 + l15;
#pragma unroll
    for (int j = 0; j < 8; ++j) {
      const long pix = pix0 + j + 8 * hf;
      x1[pix * kC + n] = x[pix * kC + n] + betaLds[n] * (acc[nt][j] + bLds[n]);
    }
  }
}

// ---------------------------------------------------------------------------
// Kernel 5: LN2 + pw3 (64->256) + SimpleGate (registers) + pw4 (128->64)
//           + residual.  Block = 2 waves; LDS ~61 KB.
// ---------------------------------------------------------------------------
__global__ __launch_bounds__(64) void k_ln2_mlp_res(
    const float* __restrict__ x1, const float* __restrict__ g2,
    const float* __restrict__ b2n, const float* __restrict__ w3,
    const float* __restrict__ b3, const float* __restrict__ w4,
    const float* __restrict__ b4, const float* __restrict__ gamma,
    float* __restrict__ out) {
  __shared__ __bf16 w3Sw[2 * 16 * 32 * 16];  // 32 KB
  __shared__ __bf16 w4Sw[4 * 4 * 32 * 16];   // 16 KB
  __shared__ float  b3Lds[kHid2];
  __shared__ __bf16 aSw[2 * 2 * 32 * 16];    //  4 KB
  __shared__ __bf16 gSw[2 * 4 * 32 * 16];    //  8 KB
  __shared__ float  b4Lds[kC], gmLds[kC];

  const int tid = threadIdx.x, wave = tid >> 5, lane = tid & 31;

  for (int i = tid; i < kC * kHid2; i += 64) {  // swizzle pw3_w
    const int kk = i >> 8, n = i & 255;
    const int t = kk >> 5, k = kk & 31;
    const int hf = (k >> 3) & 1, e = ((k >> 4) << 3) | (k & 7);
    w3Sw[((t * 16 + (n >> 4)) * 32 + hf * 16 + (n & 15)) * 16 + e] = (__bf16)w3[i];
  }
  for (int i = tid; i < kHid * kC; i += 64) {  // swizzle pw4_w
    const int kk = i >> 6, n = i & 63;
    const int t = kk >> 5, k = kk & 31;
    const int hf = (k >> 3) & 1, e = ((k >> 4) << 3) | (k & 7);
    w4Sw[((t * 4 + (n >> 4)) * 32 + hf * 16 + (n & 15)) * 16 + e] = (__bf16)w4[i];
  }
  for (int i = tid; i < kHid2; i += 64) b3Lds[i] = b3[i];
  if (tid < kC) { b4Lds[tid] = b4[tid]; gmLds[tid] = gamma[tid]; }

  const long pix0 = ((long)blockIdx.x * 2 + wave) * 16;
  __builtin_prefetch(x1 + pix0 * kC, 0, 3);

  {  // LN2 -> A fragments
    const int p = lane >> 1, c0 = (lane & 1) * 32;
    const float* xp = x1 + (pix0 + p) * kC + c0;
    float v[32], s = 0.f, q = 0.f;
#pragma unroll
    for (int c = 0; c < 32; ++c) { float t = xp[c]; v[c] = t; s += t; q += t * t; }
    s += __shfl_xor(s, 1, 32);
    q += __shfl_xor(q, 1, 32);
    const float mean = s * (1.f / kC);
    const float rstd = rsqrtf(q * (1.f / kC) - mean * mean + kEps);
#pragma unroll
    for (int c = 0; c < 32; c += 2) {
      const int cg = c0 + c;
      const float f0 = (v[c] - mean) * rstd * g2[cg] + b2n[cg];
      const float f1 = (v[c + 1] - mean) * rstd * g2[cg + 1] + b2n[cg + 1];
      const int t = cg >> 5, k = cg & 31;
      const int hf = (k >> 3) & 1, e = ((k >> 4) << 3) | (k & 7);
      *(unsigned*)&aSw[((wave * 2 + t) * 32 + hf * 16 + p) * 16 + e] = pack2(f0, f1);
    }
  }
  __syncthreads();

  const int hf = lane >> 4, l15 = lane & 15;

  const v16bf a0 = *(const v16bf*)&aSw[((wave * 2 + 0) * 32 + lane) * 16];
  const v16bf a1 = *(const v16bf*)&aSw[((wave * 2 + 1) * 32 + lane) * 16];

  v8f acc[16];
#pragma unroll
  for (int nt = 0; nt < 16; ++nt)
#pragma unroll
    for (int j = 0; j < 8; ++j) acc[nt][j] = 0.f;

#pragma unroll
  for (int nt = 0; nt < 16; ++nt) {
    const v16bf b0 = *(const v16bf*)&w3Sw[((0 * 16 + nt) * 32 + lane) * 16];
    acc[nt] = __builtin_amdgcn_wmma_f32_16x16x32_bf16(
        false, a0, false, b0, (short)0, acc[nt], false, false);
    const v16bf b1f = *(const v16bf*)&w3Sw[((1 * 16 + nt) * 32 + lane) * 16];
    acc[nt] = __builtin_amdgcn_wmma_f32_16x16x32_bf16(
        false, a1, false, b1f, (short)0, acc[nt], false, false);
  }

  // SimpleGate: n pairs with n+128 (same lane, same acc slot j, nt vs nt+8);
  // write gated value as swizzled A fragment for the pw4 GEMM.
#pragma unroll
  for (int nt = 0; nt < 8; ++nt) {
    const int n_lo = nt * 16 + l15;
    const float bl = b3Lds[n_lo], bh = b3Lds[n_lo + 128];
    const int t2 = n_lo >> 5, k2 = n_lo & 31;
    const int h2 = (k2 >> 3) & 1, e2 = ((k2 >> 4) << 3) | (k2 & 7);
#pragma unroll
    for (int j = 0; j < 8; ++j) {
      const int m = j + 8 * hf;
      gSw[((wave * 4 + t2) * 32 + h2 * 16 + m) * 16 + e2] =
          (__bf16)((acc[nt][j] + bl) * (acc[nt + 8][j] + bh));
    }
  }
  __syncthreads();

  v8f acc2[4];
#pragma unroll
  for (int nt = 0; nt < 4; ++nt)
#pragma unroll
    for (int j = 0; j < 8; ++j) acc2[nt][j] = 0.f;

#pragma unroll
  for (int t = 0; t < 4; ++t) {
    const v16bf a2 = *(const v16bf*)&gSw[((wave * 4 + t) * 32 + lane) * 16];
#pragma unroll
    for (int nt = 0; nt < 4; ++nt) {
      const v16bf bf = *(const v16bf*)&w4Sw[((t * 4 + nt) * 32 + lane) * 16];
      acc2[nt] = __builtin_amdgcn_wmma_f32_16x16x32_bf16(
          false, a2, false, bf, (short)0, acc2[nt], false, false);
    }
  }

#pragma unroll
  for (int nt = 0; nt < 4; ++nt) {
    const int n = nt * 16 + l15;
#pragma unroll
    for (int j = 0; j < 8; ++j) {
      const long pix = pix0 + j + 8 * hf;
      out[pix * kC + n] = x1[pix * kC + n] + gmLds[n] * (acc2[nt][j] + b4Lds[n]);
    }
  }
}

// ---------------------------------------------------------------------------
extern "C" void kernel_launch(void* const* d_in, const int* in_sizes, int n_in,
                              void* d_out, int out_size, void* d_ws, size_t ws_size,
                              hipStream_t stream) {
  (void)in_sizes; (void)n_in; (void)out_size; (void)ws_size;
  const float* x     = (const float*)d_in[0];
  const float* ln1_g = (const float*)d_in[1];
  const float* ln1_b = (const float*)d_in[2];
  const float* pw1_w = (const float*)d_in[3];
  const float* pw1_b = (const float*)d_in[4];
  const float* dw_w  = (const float*)d_in[5];
  const float* dw_b  = (const float*)d_in[6];
  const float* sca_w = (const float*)d_in[7];
  const float* sca_b = (const float*)d_in[8];
  const float* pw2_w = (const float*)d_in[9];
  const float* pw2_b = (const float*)d_in[10];
  const float* beta  = (const float*)d_in[11];
  const float* ln2_g = (const float*)d_in[12];
  const float* ln2_b = (const float*)d_in[13];
  const float* pw3_w = (const float*)d_in[14];
  const float* pw3_b = (const float*)d_in[15];
  const float* pw4_w = (const float*)d_in[16];
  const float* pw4_b = (const float*)d_in[17];
  const float* gamma = (const float*)d_in[18];
  float* out = (float*)d_out;

  // workspace: t1T (bf16, tiled)  268,435,456 B
  //            y   (bf16)         134,217,728 B
  //            x1  (f32)          134,217,728 B
  //            gap, wbuf (f32)    small
  char* ws = (char*)d_ws;
  __bf16* t1T  = (__bf16*)ws;
  __bf16* yb   = (__bf16*)(ws + (size_t)268435456);
  float*  x1   = (float*)(ws + (size_t)268435456 + (size_t)134217728);
  float*  gap  = (float*)(ws + (size_t)536870912);
  float*  wbuf = gap + kB * kHid;

  k_zero<<<4, 256, 0, stream>>>(gap, kB * kHid);
  k_ln1_pw1<<<4096, 256, 0, stream>>>(x, ln1_g, ln1_b, pw1_w, pw1_b, t1T);
  k_dw_gate_gap<<<2048, 256, 0, stream>>>(t1T, dw_w, dw_b, yb, gap);
  k_sca<<<1, 128, 0, stream>>>(gap, sca_w, sca_b, wbuf);
  k_attn_pw2_res<<<4096, 256, 0, stream>>>(yb, wbuf, pw2_w, pw2_b, x, beta, x1);
  k_ln2_mlp_res<<<16384, 64, 0, stream>>>(x1, ln2_g, ln2_b, pw3_w, pw3_b,
                                          pw4_w, pw4_b, gamma, out);
}
